// NodeClassifier_39711267619039
// MI455X (gfx1250) — matrix operand order repaired and checked by reading, not compiled, not run on hardware
//
#include <hip/hip_runtime.h>
#include <hip/hip_bf16.h>
#include <math.h>

#define S      64
#define NSUB   1024
#define ESUB   16384
#define NGLOB  65536
#define EGLOB  1048576
#define FIN    128
#define NHID   128
#define NCLS   64
#define KPOOL  512
#define D2     256   // 2*NHID
#define D3     384   // 3*NHID

typedef __attribute__((ext_vector_type(16))) _Float16     v16h;
typedef __attribute__((ext_vector_type(8)))  float        v8f;
typedef __attribute__((ext_vector_type(4)))  unsigned int v4u;
typedef __attribute__((ext_vector_type(8)))  int          v8i;
typedef __attribute__((ext_vector_type(4)))  int          v4i;

// ---------------------------------------------------------------------------
// TDM: DMA a 32-row x N-col f32 slice of B (row-major, stride N) into LDS.
// D# built per CDNA5 ISA 8.3/8.4:
//   group0: count=1(valid), lds_addr, global_addr[56:0], type=2
//   group1: data_size=4B(code 2), tensor_dim0=N, tensor_dim1=K,
//           tile_dim0=N, tile_dim1=32, tile_dim2=0(unused), dim0_stride=N
//   groups 2/3: zero (higher dims unused)
// ---------------------------------------------------------------------------
__device__ __forceinline__ void tdm_load_b_slice(unsigned lds_off,
                                                 const float* gptr,
                                                 int N_, int Kdim) {
  unsigned long long ga = (unsigned long long)(size_t)gptr;
  v4u g0;
  g0.x = 1u;                                            // count=1, valid
  g0.y = lds_off;                                       // LDS byte address
  g0.z = (unsigned)ga;                                  // global_addr[31:0]
  g0.w = (unsigned)((ga >> 32) & 0x1FFFFFFu) | (2u << 30); // [56:32] | type=2
  v8i g1;
  g1[0] = (int)(2u << 16);                              // data_size=4B
  g1[1] = (int)(((unsigned)N_ & 0xFFFFu) << 16);        // tensor_dim0 lo16
  g1[2] = (int)((((unsigned)N_ >> 16) & 0xFFFFu) |
                (((unsigned)Kdim & 0xFFFFu) << 16));    // dim0 hi | dim1 lo
  g1[3] = (int)((((unsigned)Kdim >> 16) & 0xFFFFu) |
                (((unsigned)N_ & 0xFFFFu) << 16));      // dim1 hi | tile_dim0=N
  g1[4] = 32;                                           // tile_dim1=32, tile_dim2=0
  g1[5] = N_;                                           // dim0_stride lo32
  g1[6] = 0;                                            // stride hi | dim1_stride lo
  g1[7] = 0;
  v4i z4 = {0, 0, 0, 0};
#if __clang_major__ >= 23
  v8i z8 = {0, 0, 0, 0, 0, 0, 0, 0};
  __builtin_amdgcn_tensor_load_to_lds(g0, g1, z4, z4, z8, 0);
#else
  __builtin_amdgcn_tensor_load_to_lds(g0, g1, z4, z4, 0);
#endif
}

// ---------------------------------------------------------------------------
// WMMA GEMM: C[M,N] = A[M,K] @ B[K,N], f32 in/out, f16 multiply, f32 accum.
// One wave per 16x16 C tile; B staged through LDS by the Tensor Data Mover,
// double-buffered (TENSORcnt) so DMA of slice c+1 overlaps compute of slice c.
// M%16==0, N%16==0 (N<=128), K%32==0.
// ---------------------------------------------------------------------------
__global__ void gemm_wmma_f16(const float* __restrict__ A,
                              const float* __restrict__ B,
                              float* __restrict__ C,
                              int M, int N_, int Kdim) {
  __shared__ float ldsB[2][32 * 128];
  int tid = threadIdx.x;
  long long gwave = ((blockIdx.x * (long long)blockDim.x + tid) >> 5);
  int lane = tid & 31;
  int tilesN = N_ >> 4;
  long long totalTiles = ((long long)(M >> 4)) * tilesN;
  bool active = gwave < totalTiles;
  int tm = active ? (int)(gwave / tilesN) : 0;
  int tn = active ? (int)(gwave % tilesN) : 0;
  int lo = lane & 15;          // row (A) / col (B,C) within tile
  int hi = lane >> 4;          // half-wave select
  int row = tm * 16 + lo;
  int col = tn * 16 + lo;
  int nChunks = Kdim >> 5;
  bool issuer = (tid < 32);    // wave 0 of the block drives the TDM

  unsigned ldsOff0 = (unsigned)(size_t)(void*)&ldsB[0][0];
  unsigned ldsOff1 = (unsigned)(size_t)(void*)&ldsB[1][0];

  if (issuer)
    tdm_load_b_slice(ldsOff0, B, N_, Kdim);

  v8f acc = {};
  for (int c = 0; c < nChunks; ++c) {
    if (issuer) {
      if (c + 1 < nChunks) {
        tdm_load_b_slice(((c + 1) & 1) ? ldsOff1 : ldsOff0,
                         B + (size_t)(c + 1) * 32 * N_, N_, Kdim);
        __builtin_amdgcn_s_wait_tensorcnt(1);   // slice c landed
      } else {
        __builtin_amdgcn_s_wait_tensorcnt(0);
      }
    }
    __syncthreads();                            // publish slice c to all waves
    if (active) {
      v16h a, b;
      // A fragment (16x32 f16): lane row=lo; elems 0..7 -> K=c*32+hi*8+i,
      // elems 8..15 -> K=c*32+16+hi*8+(i-8)   (ISA 16-bit A layout)
      const float* arow = A + (size_t)row * Kdim + c * 32 + hi * 8;
#pragma unroll
      for (int i = 0; i < 8; ++i) a[i]     = (_Float16)arow[i];
#pragma unroll
      for (int i = 0; i < 8; ++i) a[8 + i] = (_Float16)arow[16 + i];
      // B fragment (32x16 f16) from LDS: lane col=lo; elem j -> K=hi*8+(j&7)+(j>>3)*16
      const float* bl = &ldsB[c & 1][0];
#pragma unroll
      for (int j = 0; j < 16; ++j)
        b[j] = (_Float16)bl[(hi * 8 + (j & 7) + (j >> 3) * 16) * N_ + col];
      acc = __builtin_amdgcn_wmma_f32_16x16x32_f16(false, a, false, b,
                                                   (short)0, acc, false, false);
    }
    __syncthreads();                            // done reading before overwrite
  }
  if (active) {
#pragma unroll
    for (int i = 0; i < 8; ++i) {
      int r = tm * 16 + i + 8 * hi;             // C/D layout: VGPR i -> row
      C[(size_t)r * N_ + col] = acc[i];
    }
  }
}

static inline int gemm_blocks(int M, int N_) {
  long long waves = ((long long)M >> 4) * (N_ >> 4);
  long long thr = waves * 32;
  return (int)((thr + 255) / 256);
}

// ---------------------------------------------------------------------------
// Elementwise / init helpers
// ---------------------------------------------------------------------------
__global__ void init_const_k(float* p, float v, long long n) {
  long long t = blockIdx.x * (long long)blockDim.x + threadIdx.x;
  if (t < n) p[t] = v;
}

__global__ void rsqrt_inplace_k(float* p, long long n) {
  long long t = blockIdx.x * (long long)blockDim.x + threadIdx.x;
  if (t < n) p[t] = rsqrtf(p[t]);
}

// ---------------------------------------------------------------------------
// Sub-graph GCN pieces (edges: sub_edge_index[s][2][ESUB])
// ---------------------------------------------------------------------------
__global__ void sub_deg_accum_k(const int* __restrict__ ei, float* deg) {
  long long t = blockIdx.x * (long long)blockDim.x + threadIdx.x;
  if (t >= (long long)S * ESUB) return;
  int s = (int)(t / ESUB), e = (int)(t % ESUB);
  int dst = ei[(size_t)s * 2 * ESUB + ESUB + e];
  atomicAdd(&deg[s * NSUB + dst], 1.0f);
}

__global__ void sub_agg128_k(const int* __restrict__ ei,
                             const float* __restrict__ h,
                             const float* __restrict__ dinv,
                             float* __restrict__ agg) {
  long long t = blockIdx.x * (long long)blockDim.x + threadIdx.x;
  long long e = t >> 5;
  if (e >= (long long)S * ESUB) return;
  int f0 = (int)(t & 31) * 4;
  int s = (int)(e / ESUB), le = (int)(e % ESUB);
  int src = ei[(size_t)s * 2 * ESUB + le];
  int dst = ei[(size_t)s * 2 * ESUB + ESUB + le];
  float coef = dinv[s * NSUB + src] * dinv[s * NSUB + dst];
  const float* hs = h + ((size_t)s * NSUB + src) * NHID + f0;
  float* ad = agg + ((size_t)s * NSUB + dst) * NHID + f0;
#pragma unroll
  for (int i = 0; i < 4; ++i) atomicAdd(ad + i, hs[i] * coef);
}

// x = (relu?) (agg + h*dinv^2 + bias)  — in place over agg
__global__ void gcn_finalize_k(float* __restrict__ agg,
                               const float* __restrict__ h,
                               const float* __restrict__ dinv,
                               const float* __restrict__ bias,
                               long long rows, int F, int relu) {
  long long t = blockIdx.x * (long long)blockDim.x + threadIdx.x;
  if (t >= rows * F) return;
  long long r = t / F; int f = (int)(t % F);
  float di = dinv[r];
  float v = agg[t] + h[t] * di * di + bias[f];
  agg[t] = relu ? fmaxf(v, 0.0f) : v;
}

// hp[n] = dot(x1[n,:], Wp[:,0])
__global__ void matvec128_k(const float* __restrict__ x,
                            const float* __restrict__ Wp,
                            float* __restrict__ hp, long long rows) {
  long long t = blockIdx.x * (long long)blockDim.x + threadIdx.x;
  if (t >= rows) return;
  const float* xr = x + t * NHID;
  float acc = 0.f;
#pragma unroll 4
  for (int i = 0; i < NHID; ++i) acc += xr[i] * Wp[i];
  hp[t] = acc;
}

__global__ void sub_agg_scalar_k(const int* __restrict__ ei,
                                 const float* __restrict__ hp,
                                 const float* __restrict__ dinv,
                                 float* __restrict__ aggs) {
  long long t = blockIdx.x * (long long)blockDim.x + threadIdx.x;
  if (t >= (long long)S * ESUB) return;
  int s = (int)(t / ESUB), e = (int)(t % ESUB);
  int src = ei[(size_t)s * 2 * ESUB + e];
  int dst = ei[(size_t)s * 2 * ESUB + ESUB + e];
  float coef = dinv[s * NSUB + src] * dinv[s * NSUB + dst];
  atomicAdd(&aggs[s * NSUB + dst], hp[s * NSUB + src] * coef);
}

__global__ void score_finalize_k(const float* aggs, const float* hp,
                                 const float* dinv, const float* bp,
                                 float* score, long long n) {
  long long t = blockIdx.x * (long long)blockDim.x + threadIdx.x;
  if (t >= n) return;
  float di = dinv[t];
  score[t] = aggs[t] + hp[t] * di * di + bp[0];
}

// ---------------------------------------------------------------------------
// Top-K (stable, descending) via bitonic sort in LDS. One block per subgraph.
// ---------------------------------------------------------------------------
__global__ void topk_block_k(const float* __restrict__ score,
                             int* __restrict__ perm,
                             float* __restrict__ tops,
                             int* __restrict__ pos) {
  __shared__ float sv[NSUB];
  __shared__ int   si[NSUB];
  int s = blockIdx.x, t = threadIdx.x;
  sv[t] = score[s * NSUB + t];
  si[t] = t;
  __syncthreads();
  for (int k = 2; k <= NSUB; k <<= 1) {
    for (int j = k >> 1; j > 0; j >>= 1) {
      int ixj = t ^ j;
      if (ixj > t) {
        float v1 = sv[t], v2 = sv[ixj];
        int   i1 = si[t], i2 = si[ixj];
        // "precede": descending score, ascending index on ties (stable top_k)
        bool precede = (v1 > v2) || (v1 == v2 && i1 < i2);
        bool asc = ((t & k) == 0);
        if (asc ? !precede : precede) {
          sv[t] = v2; sv[ixj] = v1; si[t] = i2; si[ixj] = i1;
        }
      }
      __syncthreads();
    }
  }
  pos[s * NSUB + t] = -1;
  __syncthreads();
  if (t < KPOOL) {
    perm[s * KPOOL + t] = si[t];
    tops[s * KPOOL + t] = sv[t];
    pos[s * NSUB + si[t]] = t;
  }
}

__global__ void gather_pool_k(const float* __restrict__ x1,
                              const int* __restrict__ perm,
                              const float* __restrict__ tops,
                              float* __restrict__ x_pool) {
  long long t = blockIdx.x * (long long)blockDim.x + threadIdx.x;
  if (t >= (long long)S * KPOOL * NHID) return;
  long long r = t / NHID; int f = (int)(t % NHID);
  int s = (int)(r / KPOOL), i = (int)(r % KPOOL);
  int p = perm[s * KPOOL + i];
  float g = tanhf(tops[s * KPOOL + i]);
  x_pool[t] = x1[((size_t)s * NSUB + p) * NHID + f] * g;
}

__global__ void pool_deg_accum_k(const int* __restrict__ ei,
                                 const int* __restrict__ pos, float* deg2) {
  long long t = blockIdx.x * (long long)blockDim.x + threadIdx.x;
  if (t >= (long long)S * ESUB) return;
  int s = (int)(t / ESUB), e = (int)(t % ESUB);
  int src = ei[(size_t)s * 2 * ESUB + e];
  int dst = ei[(size_t)s * 2 * ESUB + ESUB + e];
  int ps = pos[s * NSUB + src], pd = pos[s * NSUB + dst];
  if (ps >= 0 && pd >= 0) atomicAdd(&deg2[s * KPOOL + pd], 1.0f);
}

__global__ void pool_agg128_k(const int* __restrict__ ei,
                              const int* __restrict__ pos,
                              const float* __restrict__ h,
                              const float* __restrict__ dinv2,
                              float* __restrict__ agg) {
  long long t = blockIdx.x * (long long)blockDim.x + threadIdx.x;
  long long e = t >> 5;
  if (e >= (long long)S * ESUB) return;
  int f0 = (int)(t & 31) * 4;
  int s = (int)(e / ESUB), le = (int)(e % ESUB);
  int src = ei[(size_t)s * 2 * ESUB + le];
  int dst = ei[(size_t)s * 2 * ESUB + ESUB + le];
  int ps = pos[s * NSUB + src], pd = pos[s * NSUB + dst];
  if (ps < 0 || pd < 0) return;   // ew2 == 0
  float coef = dinv2[s * KPOOL + ps] * dinv2[s * KPOOL + pd];
  const float* hs = h + ((size_t)s * KPOOL + ps) * NHID + f0;
  float* ad = agg + ((size_t)s * KPOOL + pd) * NHID + f0;
#pragma unroll
  for (int i = 0; i < 4; ++i) atomicAdd(ad + i, hs[i] * coef);
}

// sub_emb[s, f]      = max_i x_pool[s,i,f] + max_i x_sub[s,i,f]
// sub_emb[s, 128+f]  = mean  x_pool        + mean  x_sub
__global__ void emb_kernel(const float* __restrict__ x_pool,
                           const float* __restrict__ x_sub,
                           float* __restrict__ sub_emb) {
  int s = blockIdx.x, f = threadIdx.x;  // 128 threads
  float mx1 = -3.4e38f, sm1 = 0.f, mx2 = -3.4e38f, sm2 = 0.f;
  for (int i = 0; i < KPOOL; ++i) {
    float a = x_pool[((size_t)s * KPOOL + i) * NHID + f];
    float b = x_sub [((size_t)s * KPOOL + i) * NHID + f];
    mx1 = fmaxf(mx1, a); sm1 += a;
    mx2 = fmaxf(mx2, b); sm2 += b;
  }
  sub_emb[s * D2 + f]        = mx1 + mx2;
  sub_emb[s * D2 + NHID + f] = sm1 * (1.0f / KPOOL) + sm2 * (1.0f / KPOOL);
}

// Attention collapses (seq_len==1 -> softmax == 1 -> o == v):
// v[s,j]   = sub_emb[s,:] @ Wqkv[:, 512+j] + bqkv[512+j]
// att[s,j] = v[s,:] @ Wo[:, j] + bo[j]
__global__ void v_kernel(const float* __restrict__ sub_emb,
                         const float* __restrict__ Wqkv,
                         const float* __restrict__ bqkv,
                         float* __restrict__ vbuf) {
  int t = blockIdx.x * blockDim.x + threadIdx.x;
  if (t >= S * D2) return;
  int s = t / D2, j = t % D2;
  float acc = bqkv[2 * D2 + j];
  const float* e = sub_emb + s * D2;
#pragma unroll 4
  for (int i = 0; i < D2; ++i) acc += e[i] * Wqkv[(size_t)i * (3 * D2) + 2 * D2 + j];
  vbuf[t] = acc;
}

__global__ void att_kernel(const float* __restrict__ vbuf,
                           const float* __restrict__ Wo,
                           const float* __restrict__ bo,
                           float* __restrict__ att) {
  int t = blockIdx.x * blockDim.x + threadIdx.x;
  if (t >= S * D2) return;
  int s = t / D2, j = t % D2;
  float acc = bo[j];
  const float* v = vbuf + s * D2;
#pragma unroll 4
  for (int i = 0; i < D2; ++i) acc += v[i] * Wo[(size_t)i * D2 + j];
  att[t] = acc;
}

// global_emb[idx, 0:128] += x2[s,n,:]; global_emb[idx, 128:384] += att[s,:]
__global__ void scatter_kernel(const float* __restrict__ x2,
                               const float* __restrict__ att,
                               const int* __restrict__ orig_idx,
                               float* __restrict__ global_emb) {
  long long t = blockIdx.x * (long long)blockDim.x + threadIdx.x;
  if (t >= (long long)S * NSUB * D3) return;
  long long flat = t / D3; int f = (int)(t % D3);
  int s = (int)(flat >> 10);
  int gid = orig_idx[flat];
  float val = (f < NHID) ? x2[flat * NHID + f] : att[s * D2 + (f - NHID)];
  atomicAdd(&global_emb[(size_t)gid * D3 + f], val);
}

// ---------------------------------------------------------------------------
// Final global GCN
// ---------------------------------------------------------------------------
__global__ void glob_deg_accum_k(const int* __restrict__ ei, float* degf) {
  long long t = blockIdx.x * (long long)blockDim.x + threadIdx.x;
  if (t >= (long long)EGLOB) return;
  atomicAdd(&degf[ei[EGLOB + t]], 1.0f);
}

__global__ void glob_agg64_k(const int* __restrict__ ei,
                             const float* __restrict__ h,
                             const float* __restrict__ dinvf,
                             float* __restrict__ agg) {
  long long t = blockIdx.x * (long long)blockDim.x + threadIdx.x;
  long long e = t >> 4;
  if (e >= (long long)EGLOB) return;
  int f0 = (int)(t & 15) * 4;
  int src = ei[e], dst = ei[EGLOB + e];
  float coef = dinvf[src] * dinvf[dst];
  const float* hs = h + (size_t)src * NCLS + f0;
  float* ad = agg + (size_t)dst * NCLS + f0;
#pragma unroll
  for (int i = 0; i < 4; ++i) atomicAdd(ad + i, hs[i] * coef);
}

__global__ void logits_kernel(const float* __restrict__ agg,
                              const float* __restrict__ h,
                              const float* __restrict__ dinvf,
                              const float* __restrict__ bf,
                              float* __restrict__ out) {
  long long n = blockIdx.x * (long long)blockDim.x + threadIdx.x;
  if (n >= NGLOB) return;
  float di = dinvf[n], d2 = di * di;
  float l[NCLS];
  float mx = -3.4e38f;
#pragma unroll 4
  for (int f = 0; f < NCLS; ++f) {
    float v = agg[n * NCLS + f] + h[n * NCLS + f] * d2 + bf[f];
    l[f] = v;
    mx = fmaxf(mx, v);
  }
  float se = 0.f;
#pragma unroll 4
  for (int f = 0; f < NCLS; ++f) se += __expf(l[f] - mx);
  float lse = mx + __logf(se);
#pragma unroll 4
  for (int f = 0; f < NCLS; ++f) out[n * NCLS + f] = l[f] - lse;
}

// ---------------------------------------------------------------------------
// Host launch
// ---------------------------------------------------------------------------
static inline int blk(long long n, int b = 256) { return (int)((n + b - 1) / b); }

extern "C" void kernel_launch(void* const* d_in, const int* in_sizes, int n_in,
                              void* d_out, int out_size, void* d_ws, size_t ws_size,
                              hipStream_t stream) {
  const float* sub_x   = (const float*)d_in[0];
  const int*   sub_ei  = (const int*)  d_in[1];
  const int*   orig    = (const int*)  d_in[2];
  const int*   g_ei    = (const int*)  d_in[3];
  const float* W1  = (const float*)d_in[4];  const float* b1  = (const float*)d_in[5];
  const float* W2  = (const float*)d_in[6];  const float* b2  = (const float*)d_in[7];
  const float* Wp  = (const float*)d_in[8];  const float* bp  = (const float*)d_in[9];
  const float* Wsc = (const float*)d_in[10]; const float* bsc = (const float*)d_in[11];
  const float* Wqkv= (const float*)d_in[12]; const float* bqkv= (const float*)d_in[13];
  const float* Wo  = (const float*)d_in[14]; const float* bo  = (const float*)d_in[15];
  const float* Wf  = (const float*)d_in[16]; const float* bf  = (const float*)d_in[17];

  float* out_logits = (float*)d_out;                        // [N, NCLS]
  float* global_emb = (float*)d_out + (size_t)NGLOB * NCLS; // [N, D3]

  // ---- workspace layout ----
  char* w = (char*)d_ws;
  const size_t SZ32 = (size_t)S * NSUB * NHID * sizeof(float); // 32 MB
  float* bufA = (float*)(w);               // h1 / h2, later x_pool + hsc
  float* bufB = (float*)(w + SZ32);        // agg1 -> x1, later aggsc -> x_sub
  float* bufC = (float*)(w + 2 * SZ32);    // agg2 -> x2, later hf + aggf
  char* cur = w + 3 * SZ32;
  float* deg_sub = (float*)cur; cur += (size_t)S * NSUB * 4;
  float* hp      = (float*)cur; cur += (size_t)S * NSUB * 4;
  float* aggs    = (float*)cur; cur += (size_t)S * NSUB * 4;
  float* score   = (float*)cur; cur += (size_t)S * NSUB * 4;
  float* tops    = (float*)cur; cur += (size_t)S * KPOOL * 4;
  int*   perm    = (int*)  cur; cur += (size_t)S * KPOOL * 4;
  int*   pos     = (int*)  cur; cur += (size_t)S * NSUB * 4;
  float* deg2    = (float*)cur; cur += (size_t)S * KPOOL * 4;
  float* sub_emb = (float*)cur; cur += (size_t)S * D2 * 4;
  float* vbuf    = (float*)cur; cur += (size_t)S * D2 * 4;
  float* attb    = (float*)cur; cur += (size_t)S * D2 * 4;
  float* degf    = (float*)cur; cur += (size_t)NGLOB * 4;

  float* x_pool = bufA;                                    // [S*K, 128] (16 MB)
  float* hsc    = (float*)((char*)bufA + SZ32 / 2);        // [S*K, 128]
  float* x_sub  = bufB;                                    // aggsc -> x_sub in place
  float* hf     = bufC;                                    // [N, 64]
  float* aggf   = (float*)((char*)bufC + SZ32 / 2);        // [N, 64]

  const long long ROWS = (long long)S * NSUB;      // 65536
  const long long PROW = (long long)S * KPOOL;     // 32768
  const long long SE   = (long long)S * ESUB;      // 1,048,576

  // ===== layer 1: h1 = sub_x @ W1 (WMMA + TDM-staged B) =====
  gemm_wmma_f16<<<gemm_blocks((int)ROWS, NHID), 256, 0, stream>>>(
      sub_x, W1, bufA, (int)ROWS, NHID, FIN);
  init_const_k<<<blk(ROWS), 256, 0, stream>>>(deg_sub, 1.0f, ROWS);
  sub_deg_accum_k<<<blk(SE), 256, 0, stream>>>(sub_ei, deg_sub);
  rsqrt_inplace_k<<<blk(ROWS), 256, 0, stream>>>(deg_sub, ROWS);
  hipMemsetAsync(bufB, 0, SZ32, stream);
  sub_agg128_k<<<blk(SE * 32), 256, 0, stream>>>(sub_ei, bufA, deg_sub, bufB);
  gcn_finalize_k<<<blk(ROWS * NHID), 256, 0, stream>>>(bufB, bufA, deg_sub, b1,
                                                       ROWS, NHID, 1);
  // ===== layer 2: x2 = relu(gcn(x1, W2)) =====
  gemm_wmma_f16<<<gemm_blocks((int)ROWS, NHID), 256, 0, stream>>>(
      bufB, W2, bufA, (int)ROWS, NHID, NHID);
  hipMemsetAsync(bufC, 0, SZ32, stream);
  sub_agg128_k<<<blk(SE * 32), 256, 0, stream>>>(sub_ei, bufA, deg_sub, bufC);
  gcn_finalize_k<<<blk(ROWS * NHID), 256, 0, stream>>>(bufC, bufA, deg_sub, b2,
                                                       ROWS, NHID, 1);
  // ===== SAGPool scorer on x1 =====
  matvec128_k<<<blk(ROWS), 256, 0, stream>>>(bufB, Wp, hp, ROWS);
  hipMemsetAsync(aggs, 0, (size_t)ROWS * 4, stream);
  sub_agg_scalar_k<<<blk(SE), 256, 0, stream>>>(sub_ei, hp, deg_sub, aggs);
  score_finalize_k<<<blk(ROWS), 256, 0, stream>>>(aggs, hp, deg_sub, bp, score, ROWS);
  topk_block_k<<<S, NSUB, 0, stream>>>(score, perm, tops, pos);
  gather_pool_k<<<blk(PROW * NHID), 256, 0, stream>>>(bufB, perm, tops, x_pool);
  // ===== pooled GCN: x_sub = relu(gcn(x_pool, Wsc)) =====
  gemm_wmma_f16<<<gemm_blocks((int)PROW, NHID), 256, 0, stream>>>(
      x_pool, Wsc, hsc, (int)PROW, NHID, NHID);
  init_const_k<<<blk(PROW), 256, 0, stream>>>(deg2, 1.0f, PROW);
  pool_deg_accum_k<<<blk(SE), 256, 0, stream>>>(sub_ei, pos, deg2);
  rsqrt_inplace_k<<<blk(PROW), 256, 0, stream>>>(deg2, PROW);
  hipMemsetAsync(x_sub, 0, SZ32 / 2, stream);
  pool_agg128_k<<<blk(SE * 32), 256, 0, stream>>>(sub_ei, pos, hsc, deg2, x_sub);
  gcn_finalize_k<<<blk(PROW * NHID), 256, 0, stream>>>(x_sub, hsc, deg2, bsc,
                                                       PROW, NHID, 1);
  // ===== subgraph embeddings + (collapsed) attention =====
  emb_kernel<<<S, NHID, 0, stream>>>(x_pool, x_sub, sub_emb);
  v_kernel<<<blk(S * D2), 256, 0, stream>>>(sub_emb, Wqkv, bqkv, vbuf);
  att_kernel<<<blk(S * D2), 256, 0, stream>>>(vbuf, Wo, bo, attb);
  // ===== scatter into global embedding (output #2) =====
  hipMemsetAsync(global_emb, 0, (size_t)NGLOB * D3 * 4, stream);
  scatter_kernel<<<blk(ROWS * D3), 256, 0, stream>>>(bufC, attb, orig, global_emb);
  // ===== final global GCN + log_softmax =====
  gemm_wmma_f16<<<gemm_blocks(NGLOB, NCLS), 256, 0, stream>>>(
      global_emb, Wf, hf, NGLOB, NCLS, D3);
  init_const_k<<<blk(NGLOB), 256, 0, stream>>>(degf, 1.0f, NGLOB);
  glob_deg_accum_k<<<blk(EGLOB), 256, 0, stream>>>(g_ei, degf);
  rsqrt_inplace_k<<<blk(NGLOB), 256, 0, stream>>>(degf, NGLOB);
  hipMemsetAsync(aggf, 0, (size_t)NGLOB * NCLS * 4, stream);
  glob_agg64_k<<<blk((long long)EGLOB * 16), 256, 0, stream>>>(g_ei, hf, degf, aggf);
  logits_kernel<<<blk(NGLOB), 256, 0, stream>>>(aggf, hf, degf, bf, out_logits);
}
